// LSTM_VAE_ATT_18863496364504
// MI455X (gfx1250) — matrix operand order repaired and checked by reading, not compiled, not run on hardware
//
#include <hip/hip_runtime.h>

#define DEVINL __device__ __forceinline__

constexpr int Bb = 256, Tt = 128, Dd = 76, DP = 96, Hh = 128, H2 = 256, H4 = 512;
constexpr int BT = Bb * Tt;

typedef __attribute__((ext_vector_type(16))) _Float16 v16h;
typedef __attribute__((ext_vector_type(8)))  _Float16 h8v;
typedef __attribute__((ext_vector_type(8)))  float    v8f;

DEVINL v8f v8f_zero() { v8f z = {0.f,0.f,0.f,0.f,0.f,0.f,0.f,0.f}; return z; }

// A fragment: 16x32 f16, act row-major [*, lda]; rows mbase..mbase+15, K = k0..k0+31.
// ISA layout: lane l holds row l%16; halves 0..7 = K k0+(l>=16?8:0)+0..7,
// halves 8..15 = K k0+16+(l>=16?8:0)+0..7  (two contiguous 16B chunks).
DEVINL v16h frag_a(const _Float16* __restrict__ act, int lda, int mbase, int k0) {
  const int lane = threadIdx.x & 31;
  const _Float16* p = act + (size_t)(mbase + (lane & 15)) * lda + (k0 + ((lane >> 4) << 3));
  union { v16h v; h8v h[2]; } u;
  u.h[0] = *(const h8v*)(p);
  u.h[1] = *(const h8v*)(p + 16);
  return u.v;
}

// B fragment: 32x16 f16 where B[k,n] = W[n,k], W row-major [N, ldw].
// lane l holds col n = nbase + l%16; halves 0..7 = K k0+(l>=16?8:0)+0..7, halves 8..15 = K k0+16+(l>=16?8:0)+0..7.
DEVINL v16h frag_b(const _Float16* __restrict__ W, int ldw, int nbase, int k0) {
  const int lane = threadIdx.x & 31;
  const _Float16* p = W + (size_t)(nbase + (lane & 15)) * ldw + (k0 + ((lane >> 4) << 3));
  union { v16h v; h8v h[2]; } u;
  u.h[0] = *(const h8v*)(p);
  u.h[1] = *(const h8v*)(p + 16);
  return u.v;
}

DEVINL v8f wmma_f16(v16h a, v16h b, v8f c) {
  return __builtin_amdgcn_wmma_f32_16x16x32_f16(false, a, false, b, (short)0, c, false, false);
}

DEVINL float sigf(float x) { return 1.0f / (1.0f + __expf(-x)); }

// ---------------- utility kernels ----------------

__global__ void cvt_pad_kernel(const float* __restrict__ src, _Float16* __restrict__ dst,
                               int R, int C, int Rp, int Cp) {
  int i = blockIdx.x * 256 + threadIdx.x;
  if (i >= Rp * Cp) return;
  int r = i / Cp, c = i % Cp;
  dst[i] = (r < R && c < C) ? (_Float16)src[(size_t)r * C + c] : (_Float16)0.0f;
}

__global__ void zero_f32_kernel(float* p, int n) {
  int i = blockIdx.x * 256 + threadIdx.x;
  if (i < n) p[i] = 0.0f;
}
__global__ void zero_f16_kernel(_Float16* p, int n) {
  int i = blockIdx.x * 256 + threadIdx.x;
  if (i < n) p[i] = (_Float16)0.0f;
}

// ---------------- encoder LSTM step ----------------
// grid (B/64, H/16, 2 dirs), block 128 (4 waves). Wave: 16 batch rows x 16 H cols, 4 gates.
__global__ void enc_step_kernel(const _Float16* __restrict__ xh,     // [B,T,DP]
                                const _Float16* __restrict__ Wih,    // [2,4H,DP]
                                const _Float16* __restrict__ Whh,    // [2,4H,H]
                                const float*    __restrict__ bias,   // [2,4H]
                                const _Float16* __restrict__ hin,    // [2,B,H]
                                _Float16*       __restrict__ hout,   // [2,B,H]
                                float*          __restrict__ c,      // [2,B,H]
                                _Float16*       __restrict__ onh,    // [B,T,2H]
                                int step) {
  const int dir  = blockIdx.z;
  const int t    = dir ? (Tt - 1 - step) : step;
  const int warp = threadIdx.x >> 5;
  const int lane = threadIdx.x & 31;
  const int mb   = blockIdx.x * 64 + warp * 16;
  const int nb   = blockIdx.y * 16;

  const _Float16* Wih_d = Wih + (size_t)dir * H4 * DP;
  const _Float16* Whh_d = Whh + (size_t)dir * H4 * Hh;
  const float*    b_d   = bias + dir * H4;
  const _Float16* xt    = xh + (size_t)t * DP;          // rows stride T*DP
  const _Float16* hp    = hin + (size_t)dir * Bb * Hh;

  v8f acc[4];
  #pragma unroll
  for (int g = 0; g < 4; ++g) acc[g] = v8f_zero();

  for (int k0 = 0; k0 < DP; k0 += 32) {
    v16h a = frag_a(xt, Tt * DP, mb, k0);
    #pragma unroll
    for (int g = 0; g < 4; ++g)
      acc[g] = wmma_f16(a, frag_b(Wih_d, DP, g * Hh + nb, k0), acc[g]);
  }
  for (int k0 = 0; k0 < Hh; k0 += 32) {
    v16h a = frag_a(hp, Hh, mb, k0);
    #pragma unroll
    for (int g = 0; g < 4; ++g)
      acc[g] = wmma_f16(a, frag_b(Whh_d, Hh, g * Hh + nb, k0), acc[g]);
  }

  const int col = nb + (lane & 15);
  const float bi = b_d[col], bf = b_d[Hh + col], bg = b_d[2 * Hh + col], bo = b_d[3 * Hh + col];
  #pragma unroll
  for (int v = 0; v < 8; ++v) {
    const int row = mb + v + ((lane >> 4) << 3);
    const size_t ci = (size_t)dir * Bb * Hh + (size_t)row * Hh + col;
    float gi = sigf(acc[0][v] + bi);
    float gf = sigf(acc[1][v] + bf);
    float gg = tanhf(acc[2][v] + bg);
    float go = sigf(acc[3][v] + bo);
    float cn = gf * c[ci] + gi * gg;
    c[ci] = cn;
    float hn = go * tanhf(cn);
    hout[ci] = (_Float16)hn;
    onh[(size_t)row * Tt * H2 + (size_t)t * H2 + dir * Hh + col] = (_Float16)hn;
  }
}

// ---------------- decoder LSTM step ----------------
// h lives in "hq" concat layout [B, 2H], dir d occupies cols [d*H, d*H+H).
__global__ void dec_step_kernel(const _Float16* __restrict__ xh,     // [B,T,DP]
                                const _Float16* __restrict__ Wih,
                                const _Float16* __restrict__ Whh,
                                const float*    __restrict__ bias,
                                const _Float16* __restrict__ hin,    // hq prev [B,2H]
                                _Float16*       __restrict__ hout,   // hq next [B,2H]
                                float*          __restrict__ c,      // [2,B,H]
                                _Float16*       __restrict__ hcat,   // [B,4H]
                                int t) {
  const int dir  = blockIdx.z;
  const int warp = threadIdx.x >> 5;
  const int lane = threadIdx.x & 31;
  const int mb   = blockIdx.x * 64 + warp * 16;
  const int nb   = blockIdx.y * 16;

  const _Float16* Wih_d = Wih + (size_t)dir * H4 * DP;
  const _Float16* Whh_d = Whh + (size_t)dir * H4 * Hh;
  const float*    b_d   = bias + dir * H4;

  v8f acc[4];
  #pragma unroll
  for (int g = 0; g < 4; ++g) acc[g] = v8f_zero();

  if (t > 0) {  // dec input at t is x[:, t-1]; t==0 input is zeros
    const _Float16* xt = xh + (size_t)(t - 1) * DP;
    for (int k0 = 0; k0 < DP; k0 += 32) {
      v16h a = frag_a(xt, Tt * DP, mb, k0);
      #pragma unroll
      for (int g = 0; g < 4; ++g)
        acc[g] = wmma_f16(a, frag_b(Wih_d, DP, g * Hh + nb, k0), acc[g]);
    }
  }
  const _Float16* hp = hin + dir * Hh;  // lda = 2H
  for (int k0 = 0; k0 < Hh; k0 += 32) {
    v16h a = frag_a(hp, H2, mb, k0);
    #pragma unroll
    for (int g = 0; g < 4; ++g)
      acc[g] = wmma_f16(a, frag_b(Whh_d, Hh, g * Hh + nb, k0), acc[g]);
  }

  const int col = nb + (lane & 15);
  const float bi = b_d[col], bf = b_d[Hh + col], bg = b_d[2 * Hh + col], bo = b_d[3 * Hh + col];
  #pragma unroll
  for (int v = 0; v < 8; ++v) {
    const int row = mb + v + ((lane >> 4) << 3);
    const size_t ci = (size_t)dir * Bb * Hh + (size_t)row * Hh + col;
    float gi = sigf(acc[0][v] + bi);
    float gf = sigf(acc[1][v] + bf);
    float gg = tanhf(acc[2][v] + bg);
    float go = sigf(acc[3][v] + bo);
    float cn = gf * c[ci] + gi * gg;
    c[ci] = cn;
    float hn = go * tanhf(cn);
    hout[(size_t)row * H2 + dir * Hh + col] = (_Float16)hn;
    hcat[(size_t)row * H4 + dir * Hh + col] = (_Float16)hn;
  }
}

// ---------------- generic GEMM: out = act @ W^T + bias (f32 out) ----------------
// grid (M/64, N/16), block 128.
__global__ void gemm_bias_f32_kernel(const _Float16* __restrict__ act, int lda,
                                     const _Float16* __restrict__ W, int ldw,
                                     const float* __restrict__ bias,
                                     float* __restrict__ out, int ldo, int K) {
  const int warp = threadIdx.x >> 5;
  const int lane = threadIdx.x & 31;
  const int mb = blockIdx.x * 64 + warp * 16;
  const int nb = blockIdx.y * 16;
  v8f acc = v8f_zero();
  for (int k0 = 0; k0 < K; k0 += 32)
    acc = wmma_f16(frag_a(act, lda, mb, k0), frag_b(W, ldw, nb, k0), acc);
  const int col = nb + (lane & 15);
  const float b = bias[col];
  #pragma unroll
  for (int v = 0; v < 8; ++v) {
    const int row = mb + v + ((lane >> 4) << 3);
    out[(size_t)row * ldo + col] = acc[v] + b;
  }
}

// ---------------- on_s = (on@Womu^T+bmu) + eps * exp(0.5*(on@Wolv^T+blv)), f16 out ----
__global__ void onsample_kernel(const _Float16* __restrict__ onh,     // [BT,2H]
                                const _Float16* __restrict__ Wmu,
                                const _Float16* __restrict__ Wlv,
                                const float* __restrict__ bmu, const float* __restrict__ blv,
                                const float* __restrict__ eps,        // [BT,2H]
                                _Float16* __restrict__ out) {
  const int warp = threadIdx.x >> 5;
  const int lane = threadIdx.x & 31;
  const int mb = blockIdx.x * 64 + warp * 16;
  const int nb = blockIdx.y * 16;
  v8f amu = v8f_zero(), alv = v8f_zero();
  for (int k0 = 0; k0 < H2; k0 += 32) {
    v16h a = frag_a(onh, H2, mb, k0);
    amu = wmma_f16(a, frag_b(Wmu, H2, nb, k0), amu);
    alv = wmma_f16(a, frag_b(Wlv, H2, nb, k0), alv);
  }
  const int col = nb + (lane & 15);
  const float bm = bmu[col], bl = blv[col];
  #pragma unroll
  for (int v = 0; v < 8; ++v) {
    const int row = mb + v + ((lane >> 4) << 3);
    const size_t i = (size_t)row * H2 + col;
    out[i] = (_Float16)((amu[v] + bm) + eps[i] * __expf(0.5f * (alv[v] + bl)));
  }
}

// ---------------- z = mu_z + eps_z*exp(0.5*lv_z) from final enc hidden, into hq ----
// grid (B/64, H/16, 2 dirs)
__global__ void z_init_kernel(const _Float16* __restrict__ hfin,   // [2,B,H]
                              const _Float16* __restrict__ Wmu,
                              const _Float16* __restrict__ Wlv,
                              const float* __restrict__ bmu, const float* __restrict__ blv,
                              const float* __restrict__ eps_z,     // [2,B,H]
                              _Float16* __restrict__ hq) {         // [B,2H]
  const int dir  = blockIdx.z;
  const int warp = threadIdx.x >> 5;
  const int lane = threadIdx.x & 31;
  const int mb = blockIdx.x * 64 + warp * 16;
  const int nb = blockIdx.y * 16;
  const _Float16* act = hfin + (size_t)dir * Bb * Hh;
  v8f amu = v8f_zero(), alv = v8f_zero();
  for (int k0 = 0; k0 < Hh; k0 += 32) {
    v16h a = frag_a(act, Hh, mb, k0);
    amu = wmma_f16(a, frag_b(Wmu, Hh, nb, k0), amu);
    alv = wmma_f16(a, frag_b(Wlv, Hh, nb, k0), alv);
  }
  const int col = nb + (lane & 15);
  const float bm = bmu[col], bl = blv[col];
  #pragma unroll
  for (int v = 0; v < 8; ++v) {
    const int row = mb + v + ((lane >> 4) << 3);
    float z = (amu[v] + bm) +
              eps_z[(size_t)dir * Bb * Hh + (size_t)row * Hh + col] * __expf(0.5f * (alv[v] + bl));
    hq[(size_t)row * H2 + dir * Hh + col] = (_Float16)z;
  }
}

// ---------------- attention scores: one wave per (b,t') ----------------
__global__ void score_kernel(const float* __restrict__ s1,    // [B,2H]
                             const float* __restrict__ W2v,   // [BT,2H]
                             const float* __restrict__ attv,  // [2H]
                             float* __restrict__ scores) {    // [B,T]
  const int warp = threadIdx.x >> 5;
  const int lane = threadIdx.x & 31;
  const int idx = blockIdx.x * 4 + warp;   // in [0, B*T)
  const int b = idx >> 7;                  // T = 128
  const float* wrow = W2v + (size_t)idx * H2;
  const float* srow = s1 + (size_t)b * H2;
  float val = 0.0f;
  #pragma unroll
  for (int k = lane; k < H2; k += 32)
    val += tanhf(srow[k] + wrow[k]) * attv[k];
  #pragma unroll
  for (int off = 16; off; off >>= 1) val += __shfl_xor(val, off, 32);
  if (lane == 0) scores[idx] = val;
}

// ---------------- softmax over T + ctx = (w * on_s).mean into hcat[:, 2H:4H] ----
__global__ void softmax_ctx_kernel(const float* __restrict__ scores,  // [B,T]
                                   const _Float16* __restrict__ onsh, // [B,T,2H]
                                   _Float16* __restrict__ hcat) {     // [B,4H]
  __shared__ float sw[Tt];
  __shared__ float red[4];
  const int b = blockIdx.x;
  const int tid = threadIdx.x;  // 128 threads, T == 128
  float s = scores[(size_t)b * Tt + tid];
  float m = s;
  #pragma unroll
  for (int off = 16; off; off >>= 1) m = fmaxf(m, __shfl_xor(m, off, 32));
  if ((tid & 31) == 0) red[tid >> 5] = m;
  __syncthreads();
  m = fmaxf(fmaxf(red[0], red[1]), fmaxf(red[2], red[3]));
  __syncthreads();
  float e = __expf(s - m);
  float su = e;
  #pragma unroll
  for (int off = 16; off; off >>= 1) su += __shfl_xor(su, off, 32);
  if ((tid & 31) == 0) red[tid >> 5] = su;
  __syncthreads();
  su = red[0] + red[1] + red[2] + red[3];
  sw[tid] = e / su;
  __syncthreads();
  float a0 = 0.0f, a1 = 0.0f;
  const _Float16* base = onsh + (size_t)b * Tt * H2;
  for (int tt = 0; tt < Tt; ++tt) {
    float wt = sw[tt];
    a0 += wt * (float)base[(size_t)tt * H2 + tid];
    a1 += wt * (float)base[(size_t)tt * H2 + tid + Tt];
  }
  hcat[(size_t)b * H4 + H2 + tid]      = (_Float16)(a0 * (1.0f / Tt));
  hcat[(size_t)b * H4 + H2 + Tt + tid] = (_Float16)(a1 * (1.0f / Tt));
}

// ---------------- output projection: d_out[:,t,:] = hcat @ Wout^T + bout ----------------
// grid (B/64, 5), N padded 76 -> 80
__global__ void outproj_kernel(const _Float16* __restrict__ hcat,   // [B,4H]
                               const _Float16* __restrict__ WoutH,  // [80,4H]
                               const float* __restrict__ bout,
                               float* __restrict__ out, int t) {
  const int warp = threadIdx.x >> 5;
  const int lane = threadIdx.x & 31;
  const int mb = blockIdx.x * 64 + warp * 16;
  const int nb = blockIdx.y * 16;
  v8f acc = v8f_zero();
  for (int k0 = 0; k0 < H4; k0 += 32)
    acc = wmma_f16(frag_a(hcat, H4, mb, k0), frag_b(WoutH, H4, nb, k0), acc);
  const int col = nb + (lane & 15);
  #pragma unroll
  for (int v = 0; v < 8; ++v) {
    const int row = mb + v + ((lane >> 4) << 3);
    if (col < Dd)
      out[(size_t)row * Tt * Dd + (size_t)t * Dd + col] = acc[v] + bout[col];
  }
}

// ================= host side =================

extern "C" void kernel_launch(void* const* d_in, const int* in_sizes, int n_in,
                              void* d_out, int out_size, void* d_ws, size_t ws_size,
                              hipStream_t stream) {
  (void)in_sizes; (void)n_in; (void)out_size; (void)ws_size;
  const float* x       = (const float*)d_in[0];
  const float* eps_z   = (const float*)d_in[1];
  const float* eps_on  = (const float*)d_in[2];
  const float* enc_Wih = (const float*)d_in[3];
  const float* enc_Whh = (const float*)d_in[4];
  const float* enc_b   = (const float*)d_in[5];
  const float* dec_Wih = (const float*)d_in[6];
  const float* dec_Whh = (const float*)d_in[7];
  const float* dec_b   = (const float*)d_in[8];
  const float* Wzmu    = (const float*)d_in[9];
  const float* bzmu    = (const float*)d_in[10];
  const float* Wzlv    = (const float*)d_in[11];
  const float* bzlv    = (const float*)d_in[12];
  const float* Womu    = (const float*)d_in[13];
  const float* bomu    = (const float*)d_in[14];
  const float* Wolv    = (const float*)d_in[15];
  const float* bolv    = (const float*)d_in[16];
  const float* att_v   = (const float*)d_in[17];
  const float* att_W1  = (const float*)d_in[18];
  const float* att_b1  = (const float*)d_in[19];
  const float* att_W2  = (const float*)d_in[20];
  const float* att_b2  = (const float*)d_in[21];
  const float* Wout    = (const float*)d_in[22];
  const float* bout    = (const float*)d_in[23];
  float* out = (float*)d_out;

  char* ws = (char*)d_ws;
  size_t off = 0;
  auto alloc = [&](size_t bytes) -> char* {
    char* p = ws + off;
    off += (bytes + 255) & ~(size_t)255;
    return p;
  };
  _Float16* xh      = (_Float16*)alloc((size_t)BT * DP * 2);
  _Float16* encWihH = (_Float16*)alloc((size_t)2 * H4 * DP * 2);
  _Float16* encWhhH = (_Float16*)alloc((size_t)2 * H4 * Hh * 2);
  _Float16* decWihH = (_Float16*)alloc((size_t)2 * H4 * DP * 2);
  _Float16* decWhhH = (_Float16*)alloc((size_t)2 * H4 * Hh * 2);
  _Float16* WzmuH   = (_Float16*)alloc((size_t)Hh * Hh * 2);
  _Float16* WzlvH   = (_Float16*)alloc((size_t)Hh * Hh * 2);
  _Float16* WomuH   = (_Float16*)alloc((size_t)H2 * H2 * 2);
  _Float16* WolvH   = (_Float16*)alloc((size_t)H2 * H2 * 2);
  _Float16* W1H     = (_Float16*)alloc((size_t)H2 * H2 * 2);
  _Float16* W2H     = (_Float16*)alloc((size_t)H2 * H2 * 2);
  _Float16* WoutH   = (_Float16*)alloc((size_t)80 * H4 * 2);
  _Float16* onh     = (_Float16*)alloc((size_t)BT * H2 * 2);
  _Float16* onsh    = (_Float16*)alloc((size_t)BT * H2 * 2);
  float*    W2v     = (float*)   alloc((size_t)BT * H2 * 4);
  _Float16* hencA   = (_Float16*)alloc((size_t)2 * Bb * Hh * 2);
  _Float16* hencB   = (_Float16*)alloc((size_t)2 * Bb * Hh * 2);
  float*    cenc    = (float*)   alloc((size_t)2 * Bb * Hh * 4);
  _Float16* hqA     = (_Float16*)alloc((size_t)Bb * H2 * 2);
  _Float16* hqB     = (_Float16*)alloc((size_t)Bb * H2 * 2);
  float*    cdec    = (float*)   alloc((size_t)2 * Bb * Hh * 4);
  _Float16* hcat    = (_Float16*)alloc((size_t)Bb * H4 * 2);
  float*    s1      = (float*)   alloc((size_t)Bb * H2 * 4);
  float*    scores  = (float*)   alloc((size_t)Bb * Tt * 4);

  auto cvt = [&](const float* s, _Float16* d, int R, int C, int Rp, int Cp) {
    int n = Rp * Cp;
    cvt_pad_kernel<<<(n + 255) / 256, 256, 0, stream>>>(s, d, R, C, Rp, Cp);
  };
  // stage f16 (K padded to multiples of 32)
  cvt(x,       xh,      BT,     Dd, BT,     DP);
  cvt(enc_Wih, encWihH, 2 * H4, Dd, 2 * H4, DP);
  cvt(enc_Whh, encWhhH, 2 * H4, Hh, 2 * H4, Hh);
  cvt(dec_Wih, decWihH, 2 * H4, Dd, 2 * H4, DP);
  cvt(dec_Whh, decWhhH, 2 * H4, Hh, 2 * H4, Hh);
  cvt(Wzmu,    WzmuH,   Hh, Hh, Hh, Hh);
  cvt(Wzlv,    WzlvH,   Hh, Hh, Hh, Hh);
  cvt(Womu,    WomuH,   H2, H2, H2, H2);
  cvt(Wolv,    WolvH,   H2, H2, H2, H2);
  cvt(att_W1,  W1H,     H2, H2, H2, H2);
  cvt(att_W2,  W2H,     H2, H2, H2, H2);
  cvt(Wout,    WoutH,   Dd, H4, 80, H4);

  zero_f16_kernel<<<(2 * Bb * Hh + 255) / 256, 256, 0, stream>>>(hencA, 2 * Bb * Hh);
  zero_f32_kernel<<<(2 * Bb * Hh + 255) / 256, 256, 0, stream>>>(cenc, 2 * Bb * Hh);
  zero_f32_kernel<<<(2 * Bb * Hh + 255) / 256, 256, 0, stream>>>(cdec, 2 * Bb * Hh);

  // ---- bidirectional encoder (dir1 runs reversed time inside same launch) ----
  for (int s = 0; s < Tt; ++s) {
    const _Float16* hin = (s & 1) ? hencB : hencA;
    _Float16*      hout = (s & 1) ? hencA : hencB;
    enc_step_kernel<<<dim3(4, 8, 2), 128, 0, stream>>>(
        xh, encWihH, encWhhH, enc_b, hin, hout, cenc, onh, s);
  }
  // after 128 steps final hidden sits in hencA

  // z -> decoder initial hidden (concat layout hqA)
  z_init_kernel<<<dim3(4, 8, 2), 128, 0, stream>>>(hencA, WzmuH, WzlvH, bzmu, bzlv, eps_z, hqA);

  // on_s = mu_on + eps_on * exp(0.5*logvar_on)   [BT,2H] f16
  onsample_kernel<<<dim3(512, 16), 128, 0, stream>>>(onh, WomuH, WolvH, bomu, bolv, eps_on, onsh);

  // W2v = on_s @ att_W2^T + att_b2   [BT,2H] f32
  gemm_bias_f32_kernel<<<dim3(512, 16), 128, 0, stream>>>(onsh, H2, W2H, H2, att_b2, W2v, H2, H2);

  // ---- decoder ----
  for (int t = 0; t < Tt; ++t) {
    const _Float16* hin = (t & 1) ? hqB : hqA;
    _Float16*      hout = (t & 1) ? hqA : hqB;
    // s1 = q @ att_W1^T + b1 (q = concat hidden from previous step)
    gemm_bias_f32_kernel<<<dim3(4, 16), 128, 0, stream>>>(hin, H2, W1H, H2, att_b1, s1, H2, H2);
    // scores[b,t'] = sum_k tanh(s1 + W2v) * att_v
    score_kernel<<<dim3(BT / 4), 128, 0, stream>>>(s1, W2v, att_v, scores);
    // softmax over t' and ctx into hcat[:, 2H:4H]
    softmax_ctx_kernel<<<dim3(Bb), 128, 0, stream>>>(scores, onsh, hcat);
    // two LSTM cells; write new h into hq(next) and hcat[:, 0:2H]
    dec_step_kernel<<<dim3(4, 8, 2), 128, 0, stream>>>(
        xh, decWihH, decWhhH, dec_b, hin, hout, cdec, hcat, t);
    // out[:, t, :] = concat(h0,h1,ctx) @ Wout^T + bout
    outproj_kernel<<<dim3(4, 5), 128, 0, stream>>>(hcat, WoutH, bout, out, t);
  }
}
